// RNNEncoder_13469017440698
// MI455X (gfx1250) — matrix operand order repaired and checked by reading, not compiled, not run on hardware
//
#include <hip/hip_runtime.h>
#include <hip/hip_bf16.h>
#include <math.h>

// ---------------------------------------------------------------------------
// RNN encoder for MI455X (gfx1250), wave32 + v_wmma_f32_16x16x32_bf16.
//   Phase 0: stream-convert emb table / W_xh / W_hh to bf16; zero h + barrier.
//   Phase 1: fused embedding-gather GEMM  x_proj[t*64+b][n] (bf16, time-major),
//            pure bf16 loads + WMMA in the inner loop (no cvt VALU).
//   Phase 2: persistent 16-WG kernel, 128KB W_hh slice in LDS (async-staged),
//            512 sequential steps, one global-atomic barrier per step,
//            h ping-pong through L2.
// ---------------------------------------------------------------------------

typedef __attribute__((ext_vector_type(16))) __bf16 v16bf;
typedef __attribute__((ext_vector_type(8)))  float  v8f;

union BFrag { v16bf v; unsigned u[8]; uint4 q[2]; };
union CFrag { v8f v; float f[8]; };

__device__ __forceinline__ unsigned short f2bf_rn(float x) {
    unsigned u = __builtin_bit_cast(unsigned, x);
    u = (u + 0x7FFFu + ((u >> 16) & 1u)) >> 16;
    return (unsigned short)u;
}
__device__ __forceinline__ unsigned pack2_rn(float lo, float hi) {
    return (unsigned)f2bf_rn(lo) | ((unsigned)f2bf_rn(hi) << 16);
}
__device__ __forceinline__ float bf2f(unsigned short h) {
    unsigned u = ((unsigned)h) << 16;
    return __builtin_bit_cast(float, u);
}

// A fragment (16x32 bf16): lane row = lane%16, u[0..7]=K k0..k0+15,
// k0 = kb + 16*(lane/16).
__device__ __forceinline__ void load_a_bf16(BFrag& fr, const unsigned short* __restrict__ rowp, int k0) {
    fr.q[0] = *reinterpret_cast<const uint4*>(rowp + k0);        // K k0..k0+7
    fr.q[1] = *reinterpret_cast<const uint4*>(rowp + k0 + 8);    // K k0+8..k0+15
}
// B fragment (32x16 bf16): lane col = lane%16, lane-half selects 16-wide K block:
// u[0..7] = K k0..k0+15 with k0 = kb + 16*(lane/16). rowp = row of W (= col of B).
__device__ __forceinline__ void load_b_bf16(BFrag& fr, const unsigned short* __restrict__ rowp, int k0) {
    fr.q[0] = *reinterpret_cast<const uint4*>(rowp + k0);
    fr.q[1] = *reinterpret_cast<const uint4*>(rowp + k0 + 8);
}
__device__ __forceinline__ v8f wmma_bf16(const BFrag& a, const BFrag& b, v8f c) {
    return __builtin_amdgcn_wmma_f32_16x16x32_bf16(false, a.v, false, b.v, (short)0, c, false, false);
}

// ---------------------------------------------------------------------------
// Phase 0: streaming f32 -> bf16 conversion (8 floats -> one uint4 per iter)
// ---------------------------------------------------------------------------
__global__ void convert_bf16_kernel(const float* __restrict__ src,
                                    unsigned short* __restrict__ dst, long n8) {
    long i = (long)blockIdx.x * blockDim.x + threadIdx.x;
    long stride = (long)gridDim.x * blockDim.x;
    for (long k = i; k < n8; k += stride) {
        const float4* s = reinterpret_cast<const float4*>(src + k * 8);
        float4 a = s[0], b = s[1];
        uint4 o;
        o.x = pack2_rn(a.x, a.y);
        o.y = pack2_rn(a.z, a.w);
        o.z = pack2_rn(b.x, b.y);
        o.w = pack2_rn(b.z, b.w);
        reinterpret_cast<uint4*>(dst)[k] = o;
    }
}

__global__ void state_init_kernel(unsigned short* __restrict__ hping, unsigned* __restrict__ bar) {
    long i = (long)blockIdx.x * blockDim.x + threadIdx.x;   // 16384 threads, one uint4 each
    reinterpret_cast<uint4*>(hping)[i] = (uint4){0u, 0u, 0u, 0u};
    if (i == 0) { bar[0] = 0u; bar[1] = 0u; }
}

// ---------------------------------------------------------------------------
// Phase 1: x_proj[t*64+b][n] = sum_e emb_bf[src[b][t]][e] * Wxh_bf[n][e] + bxh[n]
// Block = 256 thr (8 waves), C-block 128x128, wave tile 64x32 (4x2 tiles).
// Inner loop: 12 b128 loads + 8 WMMA, no conversion VALU.
// ---------------------------------------------------------------------------
__global__ __launch_bounds__(256) void xproj_kernel(
    const int* __restrict__ src,                  // (64,512)
    const unsigned short* __restrict__ emb_bf,    // (32000,1024) bf16
    const float* __restrict__ bxh,                // (1024)
    const unsigned short* __restrict__ Wxh_bf,    // (1024,1024) bf16
    unsigned short* __restrict__ xproj)           // (32768,1024) bf16, row m = t*64+b
{
    const int lane  = threadIdx.x & 31;
    const int wave  = threadIdx.x >> 5;        // 0..7
    const int lrow  = lane & 15;
    const int khalf = lane >> 4;
    const int m0 = blockIdx.x * 128 + (wave & 1) * 64;   // wave M origin
    const int n0 = blockIdx.y * 128 + (wave >> 1) * 32;  // wave N origin

    const unsigned short* ap[4];
#pragma unroll
    for (int i = 0; i < 4; ++i) {
        int m = m0 + 16 * i + lrow;             // time-major row: b = m&63, t = m>>6
        int tok = src[(m & 63) * 512 + (m >> 6)];
        ap[i] = emb_bf + (long)tok * 1024;
    }
    const unsigned short* bp[2];
#pragma unroll
    for (int j = 0; j < 2; ++j)
        bp[j] = Wxh_bf + (long)(n0 + 16 * j + lrow) * 1024;

    v8f acc[4][2];
#pragma unroll
    for (int i = 0; i < 4; ++i)
#pragma unroll
        for (int j = 0; j < 2; ++j)
            acc[i][j] = (v8f){0.f, 0.f, 0.f, 0.f, 0.f, 0.f, 0.f, 0.f};

    for (int kb = 0; kb < 1024; kb += 32) {
        BFrag a[4], b[2];
        const int ka  = kb + khalf * 16;
        const int kbb = kb + khalf * 16;
#pragma unroll
        for (int i = 0; i < 4; ++i) load_a_bf16(a[i], ap[i], ka);
#pragma unroll
        for (int j = 0; j < 2; ++j) load_b_bf16(b[j], bp[j], kbb);
#pragma unroll
        for (int i = 0; i < 4; ++i)
#pragma unroll
            for (int j = 0; j < 2; ++j)
                acc[i][j] = wmma_bf16(a[i], b[j], acc[i][j]);
    }

#pragma unroll
    for (int j = 0; j < 2; ++j) {
        const int n = n0 + 16 * j + lrow;       // lane's fixed output column
        const float bias = bxh[n];
#pragma unroll
        for (int i = 0; i < 4; ++i) {
            CFrag c; c.v = acc[i][j];
#pragma unroll
            for (int e = 0; e < 8; ++e) {
                const int m = m0 + 16 * i + e + 8 * khalf;
                xproj[(long)m * 1024 + n] = f2bf_rn(c.f[e] + bias);
            }
        }
    }
}

// ---------------------------------------------------------------------------
// Phase 2: persistent scan. 16 WGs x 256 thr; WG owns cols 64*wg..+63,
// W_hh slice (64x1024 bf16 = 128KB) in dynamic LDS, async-staged.
// Wave = 2 C tiles sharing one A fragment: 1 A + 2 B(LDS) + 2 WMMA per K-iter.
// One barrier per step; h ping-pong (write buf != read buf => single barrier).
// ---------------------------------------------------------------------------
#define NWG2 16

__global__ __launch_bounds__(256) void rnn_scan_kernel(
    const unsigned short* __restrict__ xproj,    // (512*64,1024) bf16, time-major
    const unsigned short* __restrict__ Whh_bf,   // (1024,1024) bf16
    unsigned short* __restrict__ hping,          // 2 x (64,1024) bf16, zeroed
    float* __restrict__ hout,                    // (64,1024) f32 output
    unsigned* __restrict__ bar)                  // [0]=arrive counter, [1]=generation
{
    extern __shared__ char smem_raw[];                    // 128 KB dynamic LDS
    unsigned short* Ws = reinterpret_cast<unsigned short*>(smem_raw);

    const int wg    = blockIdx.x;
    const int lane  = threadIdx.x & 31;
    const int wave  = threadIdx.x >> 5;          // 0..7
    const int lrow  = lane & 15;
    const int khalf = lane >> 4;
    const int r  = wave & 3;                     // batch tile row 0..3
    const int c2 = wave >> 2;                    // 32-col half 0..1

    // async-stage W_hh rows n = 64*wg .. +63 (128 KB) into LDS:
    // per iteration each of 256 lanes moves 16B (ASYNCcnt-tracked DMA-style copy)
    {
        const char* gsrc = reinterpret_cast<const char*>(Whh_bf + (size_t)wg * 64 * 1024)
                         + threadIdx.x * 16;
        unsigned loff = (unsigned)(uintptr_t)smem_raw + threadIdx.x * 16;
#pragma unroll 4
        for (int it = 0; it < 32; ++it) {
            asm volatile("global_load_async_to_lds_b128 %0, %1, off"
                         :: "v"(loff), "v"(gsrc) : "memory");
            loff += 4096;
            gsrc += 4096;
        }
        asm volatile("s_wait_asynccnt 0x0" ::: "memory");
    }
    __syncthreads();

    const int nl0 = 32 * c2;                                  // wave's local col base
    const unsigned short* wrow0 = Ws + (nl0 + lrow) * 1024;   // B rows (LDS)
    const unsigned short* wrow1 = Ws + (nl0 + 16 + lrow) * 1024;
    const int n0 = 64 * wg + nl0 + lrow;                      // global output cols
    const int n1 = n0 + 16;

    for (int t = 0; t < 512; ++t) {
        const unsigned short* hread = hping + (size_t)(t & 1) * (64 * 1024);
        unsigned short* hwrite      = hping + (size_t)((t + 1) & 1) * (64 * 1024);
        const unsigned short* hrow  = hread + (16 * r + lrow) * 1024;

        v8f acc0 = (v8f){0.f, 0.f, 0.f, 0.f, 0.f, 0.f, 0.f, 0.f};
        v8f acc1 = acc0;
#pragma unroll 4
        for (int kb = 0; kb < 1024; kb += 32) {
            BFrag a, b0, b1;
            load_a_bf16(a, hrow, kb + khalf * 16);     // h tile from global (L2-hot)
            load_b_bf16(b0, wrow0, kb + khalf * 16);   // W tiles from LDS (ds_load_b128)
            load_b_bf16(b1, wrow1, kb + khalf * 16);
            acc0 = wmma_bf16(a, b0, acc0);
            acc1 = wmma_bf16(a, b1, acc1);
        }

        // epilogue: h_new = tanh(acc + x_t); write bf16 ping buffer (+f32 at t=511)
        const unsigned short* xp = xproj + (size_t)t * 64 * 1024;
        CFrag c0, c1; c0.v = acc0; c1.v = acc1;
#pragma unroll
        for (int e = 0; e < 8; ++e) {
            const int b_ = 16 * r + e + 8 * khalf;     // batch row of this element
            const float v0 = tanhf(c0.f[e] + bf2f(xp[(long)b_ * 1024 + n0]));
            const float v1 = tanhf(c1.f[e] + bf2f(xp[(long)b_ * 1024 + n1]));
            hwrite[(long)b_ * 1024 + n0] = f2bf_rn(v0);
            hwrite[(long)b_ * 1024 + n1] = f2bf_rn(v1);
            if (t == 511) {
                hout[(long)b_ * 1024 + n0] = v0;
                hout[(long)b_ * 1024 + n1] = v1;
            }
        }

        // device-wide barrier: monotonic counter + generation flag
        __threadfence();
        __syncthreads();
        if (threadIdx.x == 0) {
            const unsigned target = (unsigned)(t + 1) * NWG2;
            const unsigned old = atomicAdd(&bar[0], 1u);
            if (old + 1u == target)
                atomicExch(&bar[1], (unsigned)(t + 1));
            while (__hip_atomic_load(&bar[1], __ATOMIC_ACQUIRE, __HIP_MEMORY_SCOPE_AGENT)
                   < (unsigned)(t + 1))
                __builtin_amdgcn_s_sleep(2);
        }
        __syncthreads();
        __threadfence();
    }
}

// ---------------------------------------------------------------------------
extern "C" void kernel_launch(void* const* d_in, const int* in_sizes, int n_in,
                              void* d_out, int out_size, void* d_ws, size_t ws_size,
                              hipStream_t stream) {
    const int*   src = (const int*)  d_in[0];   // (64,512)
    const float* emb = (const float*)d_in[1];   // (32000,1024)
    const float* Wxh = (const float*)d_in[2];   // (1024,1024)
    const float* bxh = (const float*)d_in[3];   // (1024)
    const float* Whh = (const float*)d_in[4];   // (1024,1024)
    float* hout = (float*)d_out;                // (64,1024)

    // workspace layout (~136 MB):
    char* ws = (char*)d_ws;
    unsigned short* emb_bf = (unsigned short*)(ws);                          // 62.5 MB
    unsigned short* Wxh_bf = (unsigned short*)(ws + (66ull << 20));          // 2 MB
    unsigned short* Whh_bf = (unsigned short*)(ws + (68ull << 20));          // 2 MB
    unsigned short* hping  = (unsigned short*)(ws + (70ull << 20));          // 256 KB (2 bufs)
    unsigned*       bar    = (unsigned*)      (ws + (70ull << 20) + (512ull << 10));
    unsigned short* xproj  = (unsigned short*)(ws + (72ull << 20));          // 64 MB

    convert_bf16_kernel<<<2048, 256, 0, stream>>>(emb, emb_bf, 32000L * 1024 / 8);
    convert_bf16_kernel<<<256, 256, 0, stream>>>(Wxh, Wxh_bf, 1024L * 1024 / 8);
    convert_bf16_kernel<<<256, 256, 0, stream>>>(Whh, Whh_bf, 1024L * 1024 / 8);
    state_init_kernel<<<64, 256, 0, stream>>>(hping, bar);   // 16384 uint4 = 256 KB

    dim3 g1(32768 / 128, 1024 / 128);            // 256 x 8 blocks
    xproj_kernel<<<g1, 256, 0, stream>>>(src, emb_bf, bxh, Wxh_bf, xproj);

    rnn_scan_kernel<<<NWG2, 256, 128 * 1024, stream>>>(xproj, Whh_bf, hping, hout, bar);
}
